// StateHyper_30966714204398
// MI455X (gfx1250) — matrix operand order repaired and checked by reading, not compiled, not run on hardware
//
#include <hip/hip_runtime.h>
#include <math.h>

typedef __attribute__((ext_vector_type(16))) _Float16 v16h;
typedef __attribute__((ext_vector_type(8)))  float    v8f;

#define HDIM  256
#define HPDIM 128
#define BATCH 1024
#define KBIG  (HDIM * HPDIM)   // 32768

// POD 16-byte chunks (HIP's int4/float4 have ctors -> avoid in unions)
struct alignas(16) I4 { int x, y, z, w; };
struct alignas(16) F4 { float x, y, z, w; };

union V16H {
    v16h v;
    _Float16 h[16];
    I4 q[2];
};
union V8H {
    _Float16 h[8];
    I4 q;
};

__device__ __forceinline__ v8f wmma_f16(v16h a, v16h b, v8f c) {
    return __builtin_amdgcn_wmma_f32_16x16x32_f16(
        false, a, false, b, (short)0, c, false, false);
}

__device__ __forceinline__ void wait_async0() {
#if __has_builtin(__builtin_amdgcn_s_wait_asynccnt)
    __builtin_amdgcn_s_wait_asynccnt(0);
#else
    asm volatile("s_wait_asynccnt 0x0" ::: "memory");
#endif
}

// one lane copies 16B global -> LDS via the async engine (ASYNCcnt)
__device__ __forceinline__ void async_copy_b128(const _Float16* g, _Float16* l) {
    unsigned lds_off = (unsigned)(unsigned long long)l;       // low 32b = LDS addr
    unsigned long long ga = (unsigned long long)g;
    asm volatile("global_load_async_to_lds_b128 %0, %1, off"
                 :: "v"(lds_off), "v"(ga) : "memory");
}

// ---------------------------------------------------------------------------
// prep: h = cc*h0 + (1-cc)*ht -> f16 ; msg -> f16
// ---------------------------------------------------------------------------
__global__ void prep_kernel(const float* __restrict__ h0,
                            const float* __restrict__ ht,
                            const float* __restrict__ msg,
                            const float* __restrict__ c,
                            _Float16* __restrict__ h16,
                            _Float16* __restrict__ msg16) {
    int i = blockIdx.x * blockDim.x + threadIdx.x;
    float cc = fminf(fmaxf(c[0], 0.0f), 1.0f);
    if (i < BATCH * HDIM) {
        h16[i]   = (_Float16)(cc * h0[i] + (1.0f - cc) * ht[i]);
        msg16[i] = (_Float16)msg[i];
    }
}

// ---------------------------------------------------------------------------
// cvt_w: fc4_w f32 -> f16 (done once per launch; 64MB read, 32MB write)
// ---------------------------------------------------------------------------
__global__ void cvt_w_kernel(const float* __restrict__ W,
                             _Float16* __restrict__ O, int n) {
    int i = (blockIdx.x * blockDim.x + threadIdx.x) * 8;
    if (i < n) {
        F4 a = *(const F4*)(W + i);
        F4 b = *(const F4*)(W + i + 4);
        V8H o;
        o.h[0] = (_Float16)a.x; o.h[1] = (_Float16)a.y;
        o.h[2] = (_Float16)a.z; o.h[3] = (_Float16)a.w;
        o.h[4] = (_Float16)b.x; o.h[5] = (_Float16)b.y;
        o.h[6] = (_Float16)b.z; o.h[7] = (_Float16)b.w;
        *(I4*)(O + i) = o.q;
    }
}

// ---------------------------------------------------------------------------
// mlp: Y[1024 x 128] = tanh( X[1024 x K] @ W[128 x K]^T ), one wave / 16 rows
// ---------------------------------------------------------------------------
__global__ void mlp_kernel(const _Float16* __restrict__ X,
                           const float*    __restrict__ W,   // row-major [128 x K]
                           _Float16*       __restrict__ Y,
                           int K) {
    const int lane  = threadIdx.x & 31;
    const int m0    = blockIdx.x * 16;
    const int mrow  = lane & 15;
    const int half  = lane >> 4;
    const int koffA = half * 16;
    const int koffB = half * 16;

    v8f acc[8];
#pragma unroll
    for (int t = 0; t < 8; ++t)
#pragma unroll
        for (int i = 0; i < 8; ++i) acc[t][i] = 0.0f;

    const _Float16* xrow = X + (size_t)(m0 + mrow) * K;

    for (int k0 = 0; k0 < K; k0 += 32) {
        V16H a;
        a.q[0] = *(const I4*)(xrow + k0 + koffA);
        a.q[1] = *(const I4*)(xrow + k0 + koffA + 8);
#pragma unroll
        for (int t = 0; t < 8; ++t) {
            const F4* w4 = (const F4*)(W + (size_t)(t * 16 + mrow) * K + k0 + koffB);
            V16H b;
#pragma unroll
            for (int j = 0; j < 4; ++j) {
                F4 f = w4[j];
                b.h[4 * j + 0] = (_Float16)f.x;
                b.h[4 * j + 1] = (_Float16)f.y;
                b.h[4 * j + 2] = (_Float16)f.z;
                b.h[4 * j + 3] = (_Float16)f.w;
            }
            acc[t] = wmma_f16(a.v, b.v, acc[t]);
        }
    }
#pragma unroll
    for (int t = 0; t < 8; ++t)
#pragma unroll
        for (int i = 0; i < 8; ++i) {
            int m = m0 + half * 8 + i;
            int n = t * 16 + mrow;
            Y[(size_t)m * HPDIM + n] = (_Float16)tanhf(acc[t][i]);
        }
}

// ---------------------------------------------------------------------------
// hyper (f16 weights + async-to-LDS double buffering):
//   GEMM [1024 x 32768] @ [32768 x 256], A[b, h*128+r] = U[b,h]*V[b,r]
// W16 column n = contiguous 32768 f16 at n*32768. Block: 8 waves,
// Mtile 128, Ntile 64, K-step 128. LDS tile 64col x 128k f16 (16KB) x2 buffers.
// ---------------------------------------------------------------------------
__global__ void hyper_async_kernel(const _Float16* __restrict__ U,
                                   const _Float16* __restrict__ V,
                                   const _Float16* __restrict__ W16,
                                   _Float16*       __restrict__ Yh,
                                   float*          __restrict__ Yf,
                                   int mode) {
    __shared__ _Float16 Bs[2][64 * 128];   // [buf][col*128 + k], 32 KB

    const int tid  = threadIdx.x;
    const int lane = tid & 31;
    const int wave = tid >> 5;
    const int m0   = blockIdx.x * 128 + wave * 16;
    const int n0   = blockIdx.y * 64;
    const int mrow = lane & 15;
    const int half = lane >> 4;
    const int koffA = half * 8;
    const int koffB = half * 8;

    const _Float16* urow = U + (size_t)(m0 + mrow) * HDIM;
    const _Float16* vrow = V + (size_t)(m0 + mrow) * HPDIM;

    // preload this lane's entire V slice: chunks at f16 index koffA + 16*j
    I4 vc[8];
#pragma unroll
    for (int j = 0; j < 8; ++j) vc[j] = *(const I4*)(vrow + koffA + 16 * j);

    v8f acc[4];
#pragma unroll
    for (int t = 0; t < 4; ++t)
#pragma unroll
        for (int i = 0; i < 8; ++i) acc[t][i] = 0.0f;

    const _Float16* wblk = W16 + (size_t)n0 * KBIG;

    // prologue: stage tile 0 into buffer 0 (4 x b128 per thread)
#pragma unroll
    for (int j = 0; j < 4; ++j) {
        int e = j * 256 + tid;
        int col = e >> 4, seg = e & 15;
        async_copy_b128(wblk + (size_t)col * KBIG + seg * 8,
                        &Bs[0][col * 128 + seg * 8]);
    }

    const int NT = KBIG / 128;   // 256 tiles
    for (int it = 0; it < NT; ++it) {
        const int k0  = it * 128;
        const int cur = it & 1;

        wait_async0();        // this wave's copies for tile `it` have landed
        __syncthreads();      // all waves landed; prior reads of other buf done

        if (it + 1 < NT) {    // stream tile it+1 while computing tile it
            const int nk = k0 + 128;
#pragma unroll
            for (int j = 0; j < 4; ++j) {
                int e = j * 256 + tid;
                int col = e >> 4, seg = e & 15;
                async_copy_b128(wblk + (size_t)col * KBIG + nk + seg * 8,
                                &Bs[cur ^ 1][col * 128 + seg * 8]);
            }
        }

        const _Float16 us = urow[it];     // hidx = k0>>7 = it, same for whole tile
#pragma unroll
        for (int c = 0; c < 4; ++c) {     // 4 K-chunks of 32
            V16H a;
            a.q[0] = vc[2 * c];
            a.q[1] = vc[2 * c + 1];
#pragma unroll
            for (int e = 0; e < 16; ++e) a.h[e] = (_Float16)(a.h[e] * us);
#pragma unroll
            for (int t = 0; t < 4; ++t) {
                const _Float16* bb = &Bs[cur][(t * 16 + mrow) * 128 + c * 32 + koffB];
                V16H b;
                b.q[0] = *(const I4*)bb;
                b.q[1] = *(const I4*)(bb + 16);
                acc[t] = wmma_f16(a.v, b.v, acc[t]);
            }
        }
    }

    if (mode == 0) {
#pragma unroll
        for (int t = 0; t < 4; ++t)
#pragma unroll
            for (int i = 0; i < 8; ++i)
                Yh[(size_t)(m0 + half * 8 + i) * HDIM + n0 + t * 16 + mrow] =
                    (_Float16)tanhf(acc[t][i]);
    } else {
#pragma unroll
        for (int t = 0; t < 4; ++t)
#pragma unroll
            for (int i = 0; i < 8; ++i)
                Yf[(size_t)(m0 + half * 8 + i) * HDIM + n0 + t * 16 + mrow] =
                    acc[t][i];
    }
}

// ---------------------------------------------------------------------------
// fallback hyper (f32 weights, sync LDS staging) -- used if ws too small
// ---------------------------------------------------------------------------
__global__ void hyper_f32_kernel(const _Float16* __restrict__ U,
                                 const _Float16* __restrict__ V,
                                 const float*    __restrict__ W,
                                 _Float16*       __restrict__ Yh,
                                 float*          __restrict__ Yf,
                                 int mode) {
    __shared__ _Float16 Bs[64 * 64];

    const int tid  = threadIdx.x;
    const int lane = tid & 31;
    const int wave = tid >> 5;
    const int m0   = blockIdx.x * 128 + wave * 16;
    const int n0   = blockIdx.y * 64;
    const int mrow = lane & 15;
    const int half = lane >> 4;
    const int koffA = half * 8;
    const int koffB = half * 8;

    const int lcol = tid >> 2;
    const int lq   = tid & 3;
    const float* wbase = W + (size_t)(n0 + lcol) * KBIG + lq * 16;
    _Float16*    lbase = &Bs[lcol * 64 + lq * 16];

    const _Float16* urow = U + (size_t)(m0 + mrow) * HDIM;
    const _Float16* vrow = V + (size_t)(m0 + mrow) * HPDIM;

    v8f acc[4];
#pragma unroll
    for (int t = 0; t < 4; ++t)
#pragma unroll
        for (int i = 0; i < 8; ++i) acc[t][i] = 0.0f;

    for (int k0 = 0; k0 < KBIG; k0 += 64) {
        __syncthreads();
        const F4* w4 = (const F4*)(wbase + k0);
        V16H t16;
#pragma unroll
        for (int j = 0; j < 4; ++j) {
            F4 f = w4[j];
            t16.h[4 * j + 0] = (_Float16)f.x;
            t16.h[4 * j + 1] = (_Float16)f.y;
            t16.h[4 * j + 2] = (_Float16)f.z;
            t16.h[4 * j + 3] = (_Float16)f.w;
        }
        *(I4*)(lbase)     = t16.q[0];
        *(I4*)(lbase + 8) = t16.q[1];
        __syncthreads();

#pragma unroll
        for (int c = 0; c < 2; ++c) {
            const int kd0  = k0 + c * 32;
            const _Float16 us = urow[kd0 >> 7];
            const int r0 = kd0 & 127;
            V16H a;
            a.q[0] = *(const I4*)(vrow + r0 + koffA);
            a.q[1] = *(const I4*)(vrow + r0 + koffA + 16);
#pragma unroll
            for (int e = 0; e < 16; ++e) a.h[e] = (_Float16)(a.h[e] * us);
#pragma unroll
            for (int t = 0; t < 4; ++t) {
                const _Float16* bb = &Bs[(t * 16 + mrow) * 64 + c * 32 + koffB];
                V16H b;
                b.q[0] = *(const I4*)bb;
                b.q[1] = *(const I4*)(bb + 16);
                acc[t] = wmma_f16(a.v, b.v, acc[t]);
            }
        }
    }

    if (mode == 0) {
#pragma unroll
        for (int t = 0; t < 4; ++t)
#pragma unroll
            for (int i = 0; i < 8; ++i)
                Yh[(size_t)(m0 + half * 8 + i) * HDIM + n0 + t * 16 + mrow] =
                    (_Float16)tanhf(acc[t][i]);
    } else {
#pragma unroll
        for (int t = 0; t < 4; ++t)
#pragma unroll
            for (int i = 0; i < 8; ++i)
                Yf[(size_t)(m0 + half * 8 + i) * HDIM + n0 + t * 16 + mrow] =
                    acc[t][i];
    }
}

// ---------------------------------------------------------------------------
extern "C" void kernel_launch(void* const* d_in, const int* in_sizes, int n_in,
                              void* d_out, int out_size, void* d_ws, size_t ws_size,
                              hipStream_t stream) {
    const float* h0  = (const float*)d_in[0];
    const float* ht  = (const float*)d_in[1];
    const float* msg = (const float*)d_in[2];
    const float* c   = (const float*)d_in[3];
    const float* fc1 = (const float*)d_in[4];
    const float* fc2 = (const float*)d_in[5];
    const float* fc3 = (const float*)d_in[6];
    const float* fc4 = (const float*)d_in[7];
    float* out = (float*)d_out;
    (void)in_sizes; (void)n_in; (void)out_size;

    const size_t NW   = (size_t)2 * HDIM * HDIM * HPDIM;   // 16,777,216 fc4 elems
    const size_t ACT  = (size_t)BATCH * (HDIM * 3 + HPDIM * 3);
    const bool   big  = ws_size >= (NW + ACT) * sizeof(_Float16);

    _Float16* base  = (_Float16*)d_ws;
    _Float16* W16   = base;                         // only if big
    _Float16* act   = big ? base + NW : base;
    _Float16* h16   = act;
    _Float16* msg16 = h16   + BATCH * HDIM;
    _Float16* x1    = msg16 + BATCH * HDIM;
    _Float16* x2    = x1    + BATCH * HPDIM;
    _Float16* x3    = x2    + BATCH * HPDIM;
    _Float16* g1o   = x3    + BATCH * HPDIM;

    prep_kernel<<<(BATCH * HDIM) / 256, 256, 0, stream>>>(h0, ht, msg, c, h16, msg16);

    mlp_kernel<<<BATCH / 16, 32, 0, stream>>>(h16, fc1, x1, HDIM);
    mlp_kernel<<<BATCH / 16, 32, 0, stream>>>(x1,  fc2, x2, HPDIM);
    mlp_kernel<<<BATCH / 16, 32, 0, stream>>>(x2,  fc3, x3, HPDIM);

    dim3 grid(BATCH / 128, HDIM / 64);
    if (big) {
        cvt_w_kernel<<<(int)(NW / 8 / 256), 256, 0, stream>>>(fc4, W16, (int)NW);
        hyper_async_kernel<<<grid, 256, 0, stream>>>(msg16, x3, W16, g1o, nullptr, 0);
        hyper_async_kernel<<<grid, 256, 0, stream>>>(
            g1o, x3, W16 + (size_t)HDIM * HDIM * HPDIM, nullptr, out, 1);
    } else {
        hyper_f32_kernel<<<grid, 256, 0, stream>>>(msg16, x3, fc4, g1o, nullptr, 0);
        hyper_f32_kernel<<<grid, 256, 0, stream>>>(
            g1o, x3, fc4 + (size_t)HDIM * HDIM * HPDIM, nullptr, out, 1);
    }
}
